// LSTMClassifier_49383533969878
// MI455X (gfx1250) — matrix operand order repaired and checked by reading, not compiled, not run on hardware
//
#include <hip/hip_runtime.h>
#include <hip/hip_bf16.h>

// ---------------------------------------------------------------------------
// LSTM classifier: B=256, T=512, D=512, H=1024, C=16, gate order i,f,g,o.
// Per-step fused GEMM: gates = [x_t | h] @ [W_ih | W_hh]^T + (b_ih + b_hh)
// bf16 WMMA (v_wmma_f32_16x16x32_bf16), f32 accumulate, f32 cell state,
// bf16 recurrent h state. If workspace permits (~149 MB), x is pre-converted
// to bf16 once so the step hot loop is pure b128-load -> WMMA (no cvt VALU).
// No barriers in the hot loop -> compiler software-pipelines loads vs WMMA.
// ---------------------------------------------------------------------------

typedef __attribute__((ext_vector_type(16))) __bf16 v16bf;
typedef __attribute__((ext_vector_type(8)))  float  v8f;

#define B_  256
#define T_  512
#define D_  512
#define H_  1024
#define G_  4096   // 4*H
#define K_  1536   // D + H
#define C_  16

// workspace layout (bytes)
#define WCAT_OFF   0ull                      // bf16 [4096][1536] = 12,582,912 B
#define BIAS_OFF   (12582912ull)             // f32  [4096]       =     16,384 B
#define HB0_OFF    (BIAS_OFF + 16384ull)     // bf16 [256][1024]  =    524,288 B
#define HB1_OFF    (HB0_OFF + 524288ull)     // bf16 [256][1024]
#define CC_OFF     (HB1_OFF + 524288ull)     // f32  [256][1024]  =  1,048,576 B
#define XB_OFF     (CC_OFF + 1048576ull)     // bf16 [256][512][512] = 134,217,728 B
#define WS_BASE    XB_OFF                    // ~14.7 MB without xb
#define WS_FULL    (XB_OFF + 134217728ull)   // ~148.9 MB with xb

union Frag {
  v16bf v;
  struct { int4 lo, hi; } q;
  __bf16 e[16];
};

// A fragment from an f32 row: elements 0..7 <- K kb0.., 8..15 <- K kb0+16..
__device__ __forceinline__ v16bf load_a_f32(const float* __restrict__ row, int kb0) {
  Frag a;
  const float4 v0 = *(const float4*)(row + kb0);
  const float4 v1 = *(const float4*)(row + kb0 + 4);
  const float4 v2 = *(const float4*)(row + kb0 + 16);
  const float4 v3 = *(const float4*)(row + kb0 + 20);
  a.e[0]  = (__bf16)v0.x; a.e[1]  = (__bf16)v0.y; a.e[2]  = (__bf16)v0.z; a.e[3]  = (__bf16)v0.w;
  a.e[4]  = (__bf16)v1.x; a.e[5]  = (__bf16)v1.y; a.e[6]  = (__bf16)v1.z; a.e[7]  = (__bf16)v1.w;
  a.e[8]  = (__bf16)v2.x; a.e[9]  = (__bf16)v2.y; a.e[10] = (__bf16)v2.z; a.e[11] = (__bf16)v2.w;
  a.e[12] = (__bf16)v3.x; a.e[13] = (__bf16)v3.y; a.e[14] = (__bf16)v3.z; a.e[15] = (__bf16)v3.w;
  return a.v;
}

// A fragment from a bf16 row: two b128 loads, no conversion.
__device__ __forceinline__ v16bf load_a_bf16(const __bf16* __restrict__ row, int kb0) {
  Frag a;
  a.q.lo = *(const int4*)(row + kb0);
  a.q.hi = *(const int4*)(row + kb0 + 16);
  return a.v;
}

// ---------------------------------------------------------------------------
// Prep: pack [W_ih | W_hh] -> bf16 Wcat, fuse biases, zero hb0/hb1/c.
// zbase points at HB0; hb0,hb1,c are contiguous (2 MB = 524288 floats).
// ---------------------------------------------------------------------------
__global__ void lstm_prep_kernel(const float* __restrict__ W_ih,
                                 const float* __restrict__ W_hh,
                                 const float* __restrict__ b_ih,
                                 const float* __restrict__ b_hh,
                                 __bf16* __restrict__ Wcat,
                                 float*  __restrict__ bias,
                                 float*  __restrict__ zbase) {
  const int gid    = blockIdx.x * blockDim.x + threadIdx.x;
  const int stride = gridDim.x * blockDim.x;

  for (int i = gid; i < G_ * K_; i += stride) {
    const int g = i / K_;
    const int k = i - g * K_;
    const float v = (k < D_) ? W_ih[g * D_ + k] : W_hh[g * H_ + (k - D_)];
    Wcat[i] = (__bf16)v;
  }
  for (int i = gid; i < G_; i += stride) bias[i] = b_ih[i] + b_hh[i];
  for (int i = gid; i < 524288; i += stride) zbase[i] = 0.0f;  // hb0|hb1|c
}

// Convert the whole x tensor to bf16 (one-time, HBM-bandwidth bound).
__global__ void x_to_bf16_kernel(const float* __restrict__ x,
                                 __bf16* __restrict__ xb) {
  const int gid    = blockIdx.x * blockDim.x + threadIdx.x;
  const int stride = gridDim.x * blockDim.x;
  const int n4     = (B_ * T_ * D_) / 4;   // 16,777,216 float4 groups
  for (int i = gid; i < n4; i += stride) {
    const float4 v = ((const float4*)x)[i];
    __bf16* d = xb + (size_t)i * 4;
    d[0] = (__bf16)v.x; d[1] = (__bf16)v.y;
    d[2] = (__bf16)v.z; d[3] = (__bf16)v.w;
  }
}

// ---------------------------------------------------------------------------
// One LSTM timestep. Grid (8, 8), 256 threads (8 independent waves).
// Block tile: 32 batch rows x 128 hidden cols (x4 gates).
// Wave wn (0..7): 32 x 16 tile per gate -> acc[2 row-frags][4 gates].
// XB = true: x pre-converted to bf16 (zero-conversion hot loop).
// ---------------------------------------------------------------------------
template <bool XB>
__global__ __launch_bounds__(256)
void lstm_step_kernel(const float*  __restrict__ xf,     // [B][T][D] f32
                      const __bf16* __restrict__ xb,     // [B][T][D] bf16
                      const __bf16* __restrict__ Wcat,   // [4H][K]  bf16
                      const float*  __restrict__ bias,   // [4H]
                      const __bf16* __restrict__ h_in,   // [B][H]   bf16
                      __bf16*       __restrict__ h_out,  // [B][H]   bf16
                      float*        __restrict__ c,      // [B][H]   f32
                      int t) {
  const int tid   = threadIdx.x;
  const int m0    = blockIdx.x * 32;           // batch tile base
  const int wid   = tid >> 5;                  // 0..7
  const int lane  = tid & 31;
  const int ncol  = blockIdx.y * 128 + wid * 16 + (lane & 15); // hidden col
  const int khalf = lane >> 4;                 // WMMA K-half select
  const int mrow  = m0 + (lane & 15);          // A frag0 row for this lane

  // A source rows (frag0 / frag1 = +16 batch rows)
  const size_t xoff = (size_t)mrow * T_ * D_ + (size_t)t * D_;
  const float*  xfr0 = xf + xoff;
  const float*  xfr1 = xfr0 + (size_t)16 * T_ * D_;
  const __bf16* xbr0 = xb + xoff;
  const __bf16* xbr1 = xbr0 + (size_t)16 * T_ * D_;
  const __bf16* hr0  = h_in + (size_t)mrow * H_;
  const __bf16* hr1  = hr0 + (size_t)16 * H_;

  // B row pointers, one per gate (rows of Wcat, contiguous in K)
  const __bf16* wp[4];
  #pragma unroll
  for (int q = 0; q < 4; ++q)
    wp[q] = Wcat + (size_t)(q * H_ + ncol) * K_ + (size_t)khalf * 8;

  v8f acc[2][4];
  #pragma unroll
  for (int f = 0; f < 2; ++f)
    #pragma unroll
    for (int q = 0; q < 4; ++q) acc[f][q] = (v8f){};

  const int kbA = khalf * 8;   // A element-0 K offset within a 32-chunk

  // ---- Phase 1: K = [0, 512) from x ----
  for (int kk = 0; kk < D_; kk += 32) {
    v16bf a0, a1;
    if constexpr (XB) {
      a0 = load_a_bf16(xbr0, kk + kbA);
      a1 = load_a_bf16(xbr1, kk + kbA);
    } else {
      a0 = load_a_f32(xfr0, kk + kbA);
      a1 = load_a_f32(xfr1, kk + kbA);
    }
    #pragma unroll
    for (int q = 0; q < 4; ++q) {
      Frag b;
      b.q.lo = *(const int4*)(wp[q] + kk);
      b.q.hi = *(const int4*)(wp[q] + kk + 16);
      acc[0][q] = __builtin_amdgcn_wmma_f32_16x16x32_bf16(
          false, a0, false, b.v, (short)0, acc[0][q], false, false);
      acc[1][q] = __builtin_amdgcn_wmma_f32_16x16x32_bf16(
          false, a1, false, b.v, (short)0, acc[1][q], false, false);
    }
  }

  // ---- Phase 2: K = [512, 1536) from h (bf16 direct) ----
  for (int kk = 0; kk < H_; kk += 32) {
    const v16bf a0 = load_a_bf16(hr0, kk + kbA);
    const v16bf a1 = load_a_bf16(hr1, kk + kbA);
    #pragma unroll
    for (int q = 0; q < 4; ++q) {
      Frag b;
      b.q.lo = *(const int4*)(wp[q] + D_ + kk);
      b.q.hi = *(const int4*)(wp[q] + D_ + kk + 16);
      acc[0][q] = __builtin_amdgcn_wmma_f32_16x16x32_bf16(
          false, a0, false, b.v, (short)0, acc[0][q], false, false);
      acc[1][q] = __builtin_amdgcn_wmma_f32_16x16x32_bf16(
          false, a1, false, b.v, (short)0, acc[1][q], false, false);
    }
  }

  // ---- Epilogue: gates -> (h_new, c_new), all in registers ----
  const float bi = bias[0 * H_ + ncol];
  const float bf = bias[1 * H_ + ncol];
  const float bg = bias[2 * H_ + ncol];
  const float bo = bias[3 * H_ + ncol];

  #pragma unroll
  for (int f = 0; f < 2; ++f) {
    #pragma unroll
    for (int r = 0; r < 8; ++r) {
      const int m = m0 + f * 16 + r + 8 * khalf;   // C/D layout: VGPR r, lane half
      const size_t idx = (size_t)m * H_ + (size_t)ncol;
      const float iv = acc[f][0][r] + bi;
      const float fv = acc[f][1][r] + bf;
      const float gv = acc[f][2][r] + bg;
      const float ov = acc[f][3][r] + bo;
      const float ig = 1.0f / (1.0f + __expf(-iv));
      const float fg = 1.0f / (1.0f + __expf(-fv));
      const float gg = tanhf(gv);
      const float og = 1.0f / (1.0f + __expf(-ov));
      const float cn = fg * c[idx] + ig * gg;
      c[idx]     = cn;
      h_out[idx] = (__bf16)(og * tanhf(cn));
    }
  }
}

// ---------------------------------------------------------------------------
// Head: out[b][cls] = sigmoid(h[b,:] . fc_w[cls,:] + fc_b[cls]).
// ---------------------------------------------------------------------------
__global__ void lstm_head_kernel(const __bf16* __restrict__ h,
                                 const float* __restrict__ fc_w,
                                 const float* __restrict__ fc_b,
                                 float* __restrict__ out) {
  const int id  = blockIdx.x * blockDim.x + threadIdx.x; // 0..4095
  const int b   = id >> 4;
  const int cls = id & 15;
  const __bf16* hr = h + (size_t)b * H_;
  const float*  wr = fc_w + (size_t)cls * H_;
  float s = 0.0f;
  #pragma unroll 4
  for (int k = 0; k < H_; ++k) s += (float)hr[k] * wr[k];
  s += fc_b[cls];
  out[id] = 1.0f / (1.0f + __expf(-s));
}

// ---------------------------------------------------------------------------
extern "C" void kernel_launch(void* const* d_in, const int* in_sizes, int n_in,
                              void* d_out, int out_size, void* d_ws, size_t ws_size,
                              hipStream_t stream) {
  const float* x    = (const float*)d_in[0];
  const float* W_ih = (const float*)d_in[1];
  const float* W_hh = (const float*)d_in[2];
  const float* b_ih = (const float*)d_in[3];
  const float* b_hh = (const float*)d_in[4];
  const float* fc_w = (const float*)d_in[5];
  const float* fc_b = (const float*)d_in[6];

  char* ws = (char*)d_ws;
  __bf16* Wcat = (__bf16*)(ws + WCAT_OFF);
  float*  bias = (float*)(ws + BIAS_OFF);
  __bf16* hb0  = (__bf16*)(ws + HB0_OFF);
  __bf16* hb1  = (__bf16*)(ws + HB1_OFF);
  float*  cbuf = (float*)(ws + CC_OFF);
  __bf16* xbuf = (__bf16*)(ws + XB_OFF);

  const bool use_xb = (ws_size >= WS_FULL);   // fixed per run -> deterministic

  lstm_prep_kernel<<<1024, 256, 0, stream>>>(W_ih, W_hh, b_ih, b_hh,
                                             Wcat, bias, (float*)hb0);
  if (use_xb) {
    x_to_bf16_kernel<<<4096, 256, 0, stream>>>(x, xbuf);
  }

  __bf16* hbuf[2] = {hb0, hb1};
  if (use_xb) {
    for (int t = 0; t < T_; ++t) {
      lstm_step_kernel<true><<<dim3(8, 8), 256, 0, stream>>>(
          x, xbuf, Wcat, bias, hbuf[t & 1], hbuf[(t + 1) & 1], cbuf, t);
    }
  } else {
    for (int t = 0; t < T_; ++t) {
      lstm_step_kernel<false><<<dim3(8, 8), 256, 0, stream>>>(
          x, xbuf, Wcat, bias, hbuf[t & 1], hbuf[(t + 1) & 1], cbuf, t);
    }
  }

  // after 512 steps the live h buffer is hbuf[512 & 1] == hb0
  lstm_head_kernel<<<16, 256, 0, stream>>>(hb0, fc_w, fc_b, (float*)d_out);
}
